// MaskedMambaAutoencoder_6923487282741
// MI455X (gfx1250) — matrix operand order repaired and checked by reading, not compiled, not run on hardware
//
#include <hip/hip_runtime.h>
#include <math.h>

#define DEV __device__ __forceinline__

typedef __attribute__((ext_vector_type(16))) _Float16 v16h;
typedef __attribute__((ext_vector_type(8)))  _Float16 h8v;
typedef __attribute__((ext_vector_type(8)))  float    v8f;

// ---- problem dims ----
#define L_   4
#define H_   256
#define DI_  512
#define N_   16
#define KC_  4
#define B_   32
#define T_   2048
#define C_   6
#define PS_  8
#define P_   256
#define D_   48
#define DPAD_ 64         // embed K padded 48 -> 64
#define PV_  64          // visible patches per batch
#define TOK_ (B_*PV_)    // 2048 encoder tokens
#define TOKP_ (B_*P_)    // 8192 decoder tokens

// ---------------- activations ----------------
DEV float act_silu(float x)     { return x / (1.0f + expf(-x)); }
DEV float act_sigmoid(float x)  { return 1.0f / (1.0f + expf(-x)); }
DEV float act_gelu(float x)     { float x3 = x*x*x; return 0.5f*x*(1.0f + tanhf(0.7978845608f*(x + 0.044715f*x3))); }
DEV float act_softplus(float x) { return (x > 20.0f) ? x : log1pf(expf(x)); }
DEV float apply_act(float v, int act) {
    if      (act == 1) return act_silu(v);
    else if (act == 2) return act_sigmoid(v);
    else if (act == 3) return act_gelu(v);
    else if (act == 4) return act_softplus(v);
    return v;
}

// =====================================================================
// LDS-staged WMMA GEMM: C[M,N] = act(A*B + bias) + resScale*Res
//   - block tile 128x64, K-step 32, 256 threads = 8 waves
//   - wave (wr,wc) owns a 32x32 register tile -> 4 wmma accumulators,
//     each A/B fragment reused twice (4 wmma : 8 ds_load_b128)
//   - A staged row-major f16, B staged TRANSPOSED f16 so all fragments
//     are contiguous ds_load_b128's; LDS row stride 40 halfs (80B) keeps
//     16B alignment and spreads a wave's 16 rows across banks
// Requirements (met after padding): M%128==0, N%16==0, K%32==0.
// =====================================================================
#define BM 128
#define BN 64
#define BK 32
#define LSTR 40   // halfs per LDS row: 32 data + 8 pad

__global__ void __launch_bounds__(256)
gemm_wmma_kernel(const float* __restrict__ A, int lda,
                 const float* __restrict__ Bm, int ldb,
                 const float* __restrict__ bias,
                 const float* __restrict__ Res, int ldr,
                 float* __restrict__ C, int ldc,
                 int M, int N, int Kd, int act, float resScale)
{
    __shared__ _Float16 As [BM * LSTR];   // [row][k]
    __shared__ _Float16 BsT[BN * LSTR];   // [col][k]

    const int tid  = threadIdx.x;
    const int wave = tid >> 5;
    const int lane = tid & 31;
    const int nblk = (N + BN - 1) / BN;
    const int bm   = blockIdx.x / nblk;
    const int bn   = blockIdx.x - bm * nblk;
    const int m0   = bm * BM;
    const int n0   = bn * BN;

    const int wr  = wave >> 1;       // 0..3: 32-row band
    const int wc  = wave & 1;        // 0..1: 32-col band
    const int r16 = lane & 15;
    const int hi  = lane >> 4;

    // staging coordinates (fully coalesced float4 quads)
    const int arow = tid >> 1;            // 0..127
    const int akc  = (tid & 1) * 16;      // 0,16
    const int bk   = tid >> 3;            // 0..31
    const int bnc  = (tid & 7) * 8;       // 0..56

    const float* Ag = A  + (size_t)(m0 + arow) * lda + akc;
    const float* Bg = Bm + (size_t)bk * ldb + n0 + bnc;
    const bool   bOK = (n0 + bnc + 8 <= N);   // chunk fully valid (N%16==0)

    v8f acc00 = {}, acc01 = {}, acc10 = {}, acc11 = {};

    for (int k0 = 0; k0 < Kd; k0 += BK) {
        if (k0 + 2*BK < Kd) {
            __builtin_prefetch(Ag + k0 + 2*BK, 0, 1);                      // global_prefetch_b8
            __builtin_prefetch(Bg + (size_t)(k0 + 2*BK) * ldb, 0, 1);
        }
        __syncthreads();
        // ---- stage A: 16 contiguous floats -> 16 f16 (2x ds_store_b128) ----
        {
            const float4 a0 = *(const float4*)(Ag + k0);
            const float4 a1 = *(const float4*)(Ag + k0 + 4);
            const float4 a2 = *(const float4*)(Ag + k0 + 8);
            const float4 a3 = *(const float4*)(Ag + k0 + 12);
            _Float16* dst = &As[arow * LSTR + akc];
            dst[0] =(_Float16)a0.x; dst[1] =(_Float16)a0.y; dst[2] =(_Float16)a0.z; dst[3] =(_Float16)a0.w;
            dst[4] =(_Float16)a1.x; dst[5] =(_Float16)a1.y; dst[6] =(_Float16)a1.z; dst[7] =(_Float16)a1.w;
            dst[8] =(_Float16)a2.x; dst[9] =(_Float16)a2.y; dst[10]=(_Float16)a2.z; dst[11]=(_Float16)a2.w;
            dst[12]=(_Float16)a3.x; dst[13]=(_Float16)a3.y; dst[14]=(_Float16)a3.z; dst[15]=(_Float16)a3.w;
        }
        // ---- stage B transposed: row k, cols bnc..bnc+7 -> BsT[col][k] ----
        {
            float4 b0 = make_float4(0.f,0.f,0.f,0.f), b1 = b0;
            if (bOK) {
                b0 = *(const float4*)(Bg + (size_t)k0 * ldb);
                b1 = *(const float4*)(Bg + (size_t)k0 * ldb + 4);
            }
            _Float16* dst = &BsT[bnc * LSTR + bk];
            dst[0*LSTR]=(_Float16)b0.x; dst[1*LSTR]=(_Float16)b0.y;
            dst[2*LSTR]=(_Float16)b0.z; dst[3*LSTR]=(_Float16)b0.w;
            dst[4*LSTR]=(_Float16)b1.x; dst[5*LSTR]=(_Float16)b1.y;
            dst[6*LSTR]=(_Float16)b1.z; dst[7*LSTR]=(_Float16)b1.w;
        }
        __syncthreads();

        // ---- fragments from LDS (aligned ds_load_b128 pairs) ----
        // A frag lane layout: lanes0-15 K={0..7,16..23}, lanes16-31 K={8..15,24..31}
        const _Float16* ap0 = &As[(wr*32      + r16) * LSTR + 8*hi];
        const _Float16* ap1 = &As[(wr*32 + 16 + r16) * LSTR + 8*hi];
        h8v a0l = *(const h8v*)(ap0);
        h8v a0h = *(const h8v*)(ap0 + 16);
        h8v a1l = *(const h8v*)(ap1);
        h8v a1h = *(const h8v*)(ap1 + 16);
        v16h af0 = __builtin_shufflevector(a0l, a0h, 0,1,2,3,4,5,6,7,8,9,10,11,12,13,14,15);
        v16h af1 = __builtin_shufflevector(a1l, a1h, 0,1,2,3,4,5,6,7,8,9,10,11,12,13,14,15);

        // B frag lane layout: lane = col, lanes0-15 K={0..7,16..23}, lanes16-31 K={8..15,24..31}
        const _Float16* bp0 = &BsT[(wc*32      + r16) * LSTR + 8*hi];
        const _Float16* bp1 = &BsT[(wc*32 + 16 + r16) * LSTR + 8*hi];
        h8v b0l = *(const h8v*)(bp0);
        h8v b0h = *(const h8v*)(bp0 + 16);
        h8v b1l = *(const h8v*)(bp1);
        h8v b1h = *(const h8v*)(bp1 + 16);
        v16h bf0 = __builtin_shufflevector(b0l, b0h, 0,1,2,3,4,5,6,7,8,9,10,11,12,13,14,15);
        v16h bf1 = __builtin_shufflevector(b1l, b1h, 0,1,2,3,4,5,6,7,8,9,10,11,12,13,14,15);

        acc00 = __builtin_amdgcn_wmma_f32_16x16x32_f16(false, af0, false, bf0, (short)0, acc00, false, false);
        acc01 = __builtin_amdgcn_wmma_f32_16x16x32_f16(false, af0, false, bf1, (short)0, acc01, false, false);
        acc10 = __builtin_amdgcn_wmma_f32_16x16x32_f16(false, af1, false, bf0, (short)0, acc10, false, false);
        acc11 = __builtin_amdgcn_wmma_f32_16x16x32_f16(false, af1, false, bf1, (short)0, acc11, false, false);
    }

    // ---- epilogue: C/D layout col = lane&15, row = r + 8*hi ----
    const int col0 = n0 + wc*32 + r16;
    const int col1 = col0 + 16;
    const float bv0 = (bias && col0 < N) ? bias[col0] : 0.0f;
    const float bv1 = (bias && col1 < N) ? bias[col1] : 0.0f;
#pragma unroll
    for (int r = 0; r < 8; ++r) {
        const int row0 = m0 + wr*32 + r + 8*hi;
        const int row1 = row0 + 16;
        if (col0 < N) {
            float v = apply_act(acc00[r] + bv0, act);
            if (Res) v += resScale * Res[(size_t)row0 * ldr + col0];
            C[(size_t)row0 * ldc + col0] = v;
            float w = apply_act(acc10[r] + bv0, act);
            if (Res) w += resScale * Res[(size_t)row1 * ldr + col0];
            C[(size_t)row1 * ldc + col0] = w;
        }
        if (col1 < N) {
            float v = apply_act(acc01[r] + bv1, act);
            if (Res) v += resScale * Res[(size_t)row0 * ldr + col1];
            C[(size_t)row0 * ldc + col1] = v;
            float w = apply_act(acc11[r] + bv1, act);
            if (Res) w += resScale * Res[(size_t)row1 * ldr + col1];
            C[(size_t)row1 * ldc + col1] = w;
        }
    }
}

// =====================================================================
// PDGM mask: gradient intensity -> softmax -> clip/renorm -> gumbel top-k.
// One block per batch row; bitonic sort of 256 scores; visible = 64 lowest.
// (JAX threefry stream is not bit-reproducible; splitmix64 stands in — the
// index path is stop_gradient / non-differentiable in the reference.)
// =====================================================================
__global__ void mask_kernel(const float* __restrict__ raw, int* __restrict__ vis)
{
    __shared__ float sc[P_];
    __shared__ int   si[P_];
    __shared__ float red[P_];
    __shared__ int   flag[P_];
    __shared__ int   pref[P_];
    const int b = blockIdx.x, p = threadIdx.x;

    float s = 0.0f;
    for (int j = 0; j < PS_; ++j) {
        int t  = p * PS_ + j;
        int tt = (t < T_ - 1) ? t : T_ - 2;
        float g2 = 0.0f;
        for (int c = 0; c < C_; ++c) {
            float d = raw[(size_t)(b*T_ + tt + 1)*C_ + c] - raw[(size_t)(b*T_ + tt)*C_ + c];
            g2 += d * d;
        }
        s += sqrtf(g2);
    }
    float inten = s * (1.0f / PS_);

    red[p] = inten; __syncthreads();
    for (int st = 128; st > 0; st >>= 1) { if (p < st) red[p] = fmaxf(red[p], red[p+st]); __syncthreads(); }
    float mx = red[0]; __syncthreads();
    float e = expf(inten - mx);
    red[p] = e; __syncthreads();
    for (int st = 128; st > 0; st >>= 1) { if (p < st) red[p] += red[p+st]; __syncthreads(); }
    float sum = red[0]; __syncthreads();
    float prob = fmaxf(e / sum, 0.01f);
    red[p] = prob; __syncthreads();
    for (int st = 128; st > 0; st >>= 1) { if (p < st) red[p] += red[p+st]; __syncthreads(); }
    float s2 = red[0]; __syncthreads();
    prob /= s2;

    unsigned long long h = (unsigned long long)(b * P_ + p) * 0x9E3779B97F4A7C15ull + 42ull;
    h ^= h >> 30; h *= 0xBF58476D1CE4E5B9ull;
    h ^= h >> 27; h *= 0x94D049BB133111EBull;
    h ^= h >> 31;
    float uu  = fmaxf((float)((h >> 40) & 0xFFFFFF) * (1.0f/16777216.0f), 1e-8f);
    float gum = -logf(-logf(uu));
    sc[p] = logf(prob + 1e-8f) + gum;
    si[p] = p;
    flag[p] = 0;
    __syncthreads();

    for (int k = 2; k <= P_; k <<= 1)
        for (int j = k >> 1; j > 0; j >>= 1) {
            int ixj = p ^ j;
            if (ixj > p) {
                bool asc = ((p & k) == 0);
                float a0 = sc[p], a1 = sc[ixj];
                if ((a0 > a1) == asc) {
                    sc[p] = a1; sc[ixj] = a0;
                    int t0 = si[p]; si[p] = si[ixj]; si[ixj] = t0;
                }
            }
            __syncthreads();
        }
    if (p < PV_) flag[si[p]] = 1;
    __syncthreads();

    pref[p] = flag[p]; __syncthreads();
    for (int st = 1; st < P_; st <<= 1) {
        int v2 = (p >= st) ? pref[p - st] : 0;
        __syncthreads();
        pref[p] += v2;
        __syncthreads();
    }
    if (flag[p]) vis[b * PV_ + (pref[p] - 1)] = p;
}

// ---------------- layernorm: one token per block (H=256 threads) --------
__global__ void ln_kernel(const float* __restrict__ x, const float* __restrict__ w,
                          const float* __restrict__ bb, float* __restrict__ y)
{
    __shared__ float red[H_];
    const int tok = blockIdx.x, h = threadIdx.x;
    float v = x[(size_t)tok * H_ + h];
    red[h] = v; __syncthreads();
    for (int st = 128; st > 0; st >>= 1) { if (h < st) red[h] += red[h+st]; __syncthreads(); }
    float mu = red[0] * (1.0f / H_); __syncthreads();
    float dv = v - mu;
    red[h] = dv * dv; __syncthreads();
    for (int st = 128; st > 0; st >>= 1) { if (h < st) red[h] += red[h+st]; __syncthreads(); }
    float var = red[0] * (1.0f / H_);
    y[(size_t)tok * H_ + h] = dv * rsqrtf(var + 1e-5f) * w[h] + bb[h];
}

// ---- SSM scan: thread = (b,d); 16 states in registers; BCs = [Bs|Cs] ----
__global__ void ssm_scan_kernel(const float* __restrict__ u,  const float* __restrict__ dtb,
                                const float* __restrict__ BCs,
                                const float* __restrict__ logA, const float* __restrict__ Dp,
                                float* __restrict__ out)
{
    const int g = blockIdx.x * blockDim.x + threadIdx.x;   // 0..B*DI-1
    const int b = g / DI_;
    const int d = g - b * DI_;        // lane-major over d -> BCs loads wave-uniform
    float A[N_], xs[N_];
#pragma unroll
    for (int n = 0; n < N_; ++n) { A[n] = -expf(logA[d * N_ + n]); xs[n] = 0.0f; }
    const float dp = Dp[d];
    for (int t = 0; t < PV_; ++t) {
        const int ti = b * PV_ + t;
        float dt = dtb[(size_t)ti * DI_ + d];
        float uu = u[(size_t)ti * DI_ + d];
        float du = dt * uu;
        float y = 0.0f;
        const float* bc = BCs + (size_t)ti * (2 * N_);
#pragma unroll
        for (int n = 0; n < N_; ++n) {
            float dA = expf(dt * A[n]);
            xs[n] = dA * xs[n] + du * bc[n];          // Bs
            y += xs[n] * bc[N_ + n];                  // Cs
        }
        out[(size_t)ti * DI_ + d] = y + uu * dp;
    }
}

// ---------------- small elementwise glue kernels ----------------
__global__ void gather_patches_kernel(const float* __restrict__ patches, const int* __restrict__ vis,
                                      float* __restrict__ visp, int total)   // DPAD_-wide, zero pad
{
    int i = blockIdx.x * blockDim.x + threadIdx.x; if (i >= total) return;
    int dd = i % DPAD_; int v = (i / DPAD_) % PV_; int b = i / (DPAD_ * PV_);
    visp[i] = (dd < D_) ? patches[((size_t)b * P_ + vis[b * PV_ + v]) * D_ + dd] : 0.0f;
}
__global__ void pad_wembed_kernel(const float* __restrict__ W, float* __restrict__ out, int total)
{
    int i = blockIdx.x * blockDim.x + threadIdx.x; if (i >= total) return;
    int r = i / H_; int c = i % H_;
    out[i] = (r < D_) ? W[(size_t)r * H_ + c] : 0.0f;
}
__global__ void pack_wbc_kernel(const float* __restrict__ Wb, const float* __restrict__ Wc,
                                float* __restrict__ out, int total)   // [DI][32] = [Wb|Wc]
{
    int i = blockIdx.x * blockDim.x + threadIdx.x; if (i >= total) return;
    int n = i % (2 * N_); int k = i / (2 * N_);
    out[i] = (n < N_) ? Wb[(size_t)k * N_ + n] : Wc[(size_t)k * N_ + (n - N_)];
}
__global__ void add_pos_kernel(float* __restrict__ x, const float* __restrict__ pos,
                               const int* __restrict__ vis, int total)
{
    int i = blockIdx.x * blockDim.x + threadIdx.x; if (i >= total) return;
    int h = i % H_; int v = (i / H_) % PV_; int b = i / (H_ * PV_);
    x[i] += pos[(size_t)vis[b * PV_ + v] * H_ + h];
}
__global__ void flip_kernel(const float* __restrict__ in, float* __restrict__ out, int width, int total)
{
    int i = blockIdx.x * blockDim.x + threadIdx.x; if (i >= total) return;
    int h = i % width; int t = (i / width) % PV_; int b = i / (width * PV_);
    out[((size_t)b * PV_ + (PV_ - 1 - t)) * width + h] = in[i];
}
__global__ void im2col_kernel(const float* __restrict__ xz, float* __restrict__ out, int total)
{
    int i = blockIdx.x * blockDim.x + threadIdx.x; if (i >= total) return;
    int d = i % DI_; int k = (i / DI_) % KC_; int t = (i / (DI_ * KC_)) % PV_; int b = i / (DI_ * KC_ * PV_);
    int ts = t - (KC_ - 1) + k;
    float v = (ts >= 0) ? xz[((size_t)b * PV_ + ts) * (2 * DI_) + d] : 0.0f;  // xb = first half of xz
    out[((size_t)b * PV_ + t) * (KC_ * DI_) + k * DI_ + d] = v;
}
__global__ void packw_kernel(const float* __restrict__ cw, float* __restrict__ out, int total)
{
    int i = blockIdx.x * blockDim.x + threadIdx.x; if (i >= total) return;   // out[(k*DI+in), o]
    int o = i % DI_; int kk = i / DI_; int k = kk / DI_; int ii = kk - k * DI_;
    out[(size_t)kk * DI_ + o] = cw[((size_t)o * DI_ + ii) * KC_ + k];
}
__global__ void gatemul_kernel(const float* __restrict__ s, const float* __restrict__ xz,
                               float* __restrict__ g, int total)
{
    int i = blockIdx.x * blockDim.x + threadIdx.x; if (i >= total) return;
    int d = i % DI_; int t = i / DI_;
    float z = xz[(size_t)t * (2 * DI_) + DI_ + d];
    g[i] = s[i] * act_silu(z);
}
__global__ void concat_kernel(const float* __restrict__ hf, const float* __restrict__ hb,
                              float* __restrict__ comb, int total)
{
    int i = blockIdx.x * blockDim.x + threadIdx.x; if (i >= total) return;
    int h = i % (2 * H_); int t = i / (2 * H_);
    comb[i] = (h < H_) ? hf[(size_t)t * H_ + h] : hb[(size_t)t * H_ + (h - H_)];
}
__global__ void combine_kernel(const float* __restrict__ g, const float* __restrict__ hf,
                               const float* __restrict__ hb, const float* __restrict__ m,
                               float* __restrict__ x, int total)
{
    int i = blockIdx.x * blockDim.x + threadIdx.x; if (i >= total) return;
    float gg = g[i];
    x[i] = gg * hf[i] + (1.0f - gg) * hb[i] + 0.1f * m[i];
}
__global__ void build_full_kernel(const float* __restrict__ mtok, const float* __restrict__ pos,
                                  float* __restrict__ full, int total)
{
    int i = blockIdx.x * blockDim.x + threadIdx.x; if (i >= total) return;
    int h = i % H_; int p = (i / H_) % P_;
    full[i] = mtok[h] + pos[(size_t)p * H_ + h];
}
__global__ void scatter_vis_kernel(const float* __restrict__ x, const int* __restrict__ vis,
                                   float* __restrict__ full, int total)
{
    int i = blockIdx.x * blockDim.x + threadIdx.x; if (i >= total) return;
    int h = i % H_; int v = (i / H_) % PV_; int b = i / (H_ * PV_);
    full[((size_t)b * P_ + vis[b * PV_ + v]) * H_ + h] = x[i];
}

// =====================================================================
// Host orchestration
// =====================================================================
extern "C" void kernel_launch(void* const* d_in, const int* in_sizes, int n_in,
                              void* d_out, int out_size, void* d_ws, size_t ws_size,
                              hipStream_t stream)
{
    const float* patches   = (const float*)d_in[0];
    const float* raw       = (const float*)d_in[1];
    const float* W_embed   = (const float*)d_in[2];
    const float* b_embed   = (const float*)d_in[3];
    const float* pos_embed = (const float*)d_in[4];
    const float* mask_tok  = (const float*)d_in[5];
    const float* ln_w      = (const float*)d_in[6];
    const float* ln_b      = (const float*)d_in[7];
    const float* W_in      = (const float*)d_in[8];
    const float* conv_w    = (const float*)d_in[9];
    const float* conv_b    = (const float*)d_in[10];
    const float* Wb        = (const float*)d_in[11];
    const float* Wc        = (const float*)d_in[12];
    const float* Wdt       = (const float*)d_in[13];
    const float* bdt       = (const float*)d_in[14];
    const float* logA      = (const float*)d_in[15];
    const float* Dp        = (const float*)d_in[16];
    const float* W_out     = (const float*)d_in[17];
    const float* W_gate    = (const float*)d_in[18];
    const float* b_gate    = (const float*)d_in[19];
    const float* W_merge   = (const float*)d_in[20];
    const float* dec_W1    = (const float*)d_in[21];
    const float* dec_b1    = (const float*)d_in[22];
    const float* dec_W2    = (const float*)d_in[23];
    const float* dec_b2    = (const float*)d_in[24];
    (void)in_sizes; (void)n_in; (void)out_size; (void)ws_size;

    char* wsb = (char*)d_ws;
    size_t off = 0;
    auto alloc = [&](size_t bytes) -> char* {
        char* p = wsb + off; off += (bytes + 255) & ~(size_t)255; return p;
    };
    int*   vis    = (int*)  alloc((size_t)B_ * PV_ * 4);
    float* x      = (float*)alloc((size_t)TOK_ * H_ * 4);
    float* visp   = (float*)alloc((size_t)TOK_ * DPAD_ * 4);
    float* wepad  = (float*)alloc((size_t)DPAD_ * H_ * 4);
    float* lnbuf  = (float*)alloc((size_t)TOK_ * H_ * 4);
    float* xz     = (float*)alloc((size_t)TOK_ * 2 * DI_ * 4);
    float* convin = (float*)alloc((size_t)TOK_ * KC_ * DI_ * 4);
    float* wpack  = (float*)alloc((size_t)KC_ * DI_ * DI_ * 4);
    float* wbc    = (float*)alloc((size_t)DI_ * 2 * N_ * 4);
    float* ubuf   = (float*)alloc((size_t)TOK_ * DI_ * 4);
    float* bcs    = (float*)alloc((size_t)TOK_ * 2 * N_ * 4);
    float* dtb    = (float*)alloc((size_t)TOK_ * DI_ * 4);
    float* sout   = (float*)alloc((size_t)TOK_ * DI_ * 4);
    float* gated  = (float*)alloc((size_t)TOK_ * DI_ * 4);
    float* hf     = (float*)alloc((size_t)TOK_ * H_ * 4);
    float* hrev   = (float*)alloc((size_t)TOK_ * H_ * 4);
    float* hb     = (float*)alloc((size_t)TOK_ * H_ * 4);
    float* xrev   = (float*)alloc((size_t)TOK_ * H_ * 4);
    float* comb   = (float*)alloc((size_t)TOK_ * 2 * H_ * 4);
    float* gbuf   = (float*)alloc((size_t)TOK_ * H_ * 4);
    float* mbuf   = (float*)alloc((size_t)TOK_ * H_ * 4);
    float* full   = (float*)alloc((size_t)TOKP_ * H_ * 4);
    float* hdec   = (float*)alloc((size_t)TOKP_ * H_ * 4);

    auto gemm = [&](const float* A, int lda, const float* Bm, int ldb, const float* bias,
                    const float* Res, int ldr, float* Cm, int ldc,
                    int M, int N, int K, int act, float resScale) {
        int blocks = (M / BM) * ((N + BN - 1) / BN);
        gemm_wmma_kernel<<<blocks, 256, 0, stream>>>(A, lda, Bm, ldb, bias, Res, ldr,
                                                     Cm, ldc, M, N, K, act, resScale);
    };
    auto ew = [&](int total) { return dim3((total + 255) / 256); };

    // ---- 1. dynamics-guided mask + visible gather + embed ----
    mask_kernel<<<B_, 256, 0, stream>>>(raw, vis);
    gather_patches_kernel<<<ew(TOK_*DPAD_), 256, 0, stream>>>(patches, vis, visp, TOK_*DPAD_);
    pad_wembed_kernel<<<ew(DPAD_*H_), 256, 0, stream>>>(W_embed, wepad, DPAD_*H_);
    gemm(visp, DPAD_, wepad, H_, b_embed, nullptr, 0, x, H_, TOK_, H_, DPAD_, 0, 0.0f);
    add_pos_kernel<<<ew(TOK_*H_), 256, 0, stream>>>(x, pos_embed, vis, TOK_*H_);

    // ---- 2. bidirectional Mamba layers ----
    for (int l = 0; l < L_; ++l) {
        flip_kernel<<<ew(TOK_*H_), 256, 0, stream>>>(x, xrev, H_, TOK_*H_);
        for (int dir = 0; dir < 2; ++dir) {
            const float* xin  = dir ? xrev : x;
            float*       hout = dir ? hrev : hf;
            const size_t po = (size_t)(l * 2 + dir);

            ln_kernel<<<TOK_, 256, 0, stream>>>(xin, ln_w + po*H_, ln_b + po*H_, lnbuf);
            gemm(lnbuf, H_, W_in + po*H_*2*DI_, 2*DI_, nullptr, nullptr, 0,
                 xz, 2*DI_, TOK_, 2*DI_, H_, 0, 0.0f);

            // causal conv as K = KC*DI GEMM
            im2col_kernel<<<ew(TOK_*KC_*DI_), 256, 0, stream>>>(xz, convin, TOK_*KC_*DI_);
            packw_kernel<<<ew(KC_*DI_*DI_), 256, 0, stream>>>(conv_w + po*DI_*DI_*KC_, wpack, KC_*DI_*DI_);
            gemm(convin, KC_*DI_, wpack, DI_, conv_b + po*DI_, nullptr, 0,
                 ubuf, DI_, TOK_, DI_, KC_*DI_, 1 /*silu*/, 0.0f);

            // SSM projections: fused [Bs|Cs] (N=32) + dt
            pack_wbc_kernel<<<ew(DI_*2*N_), 256, 0, stream>>>(Wb + po*DI_*N_, Wc + po*DI_*N_,
                                                              wbc, DI_*2*N_);
            gemm(ubuf, DI_, wbc, 2*N_, nullptr, nullptr, 0, bcs, 2*N_, TOK_, 2*N_, DI_, 0, 0.0f);
            gemm(ubuf, DI_, Wdt + po*DI_*DI_, DI_, bdt + po*DI_, nullptr, 0,
                 dtb, DI_, TOK_, DI_, DI_, 4 /*softplus*/, 0.0f);

            ssm_scan_kernel<<<(B_*DI_)/256, 256, 0, stream>>>(ubuf, dtb, bcs,
                                                              logA + po*DI_*N_, Dp + po*DI_, sout);
            gatemul_kernel<<<ew(TOK_*DI_), 256, 0, stream>>>(sout, xz, gated, TOK_*DI_);
            gemm(gated, DI_, W_out + po*DI_*H_, H_, nullptr, xin, H_,
                 hout, H_, TOK_, H_, DI_, 0, 1.0f);   // + residual
        }
        flip_kernel<<<ew(TOK_*H_), 256, 0, stream>>>(hrev, hb, H_, TOK_*H_);

        concat_kernel<<<ew(TOK_*2*H_), 256, 0, stream>>>(hf, hb, comb, TOK_*2*H_);
        gemm(comb, 2*H_, W_gate  + (size_t)l*2*H_*H_, H_, b_gate + (size_t)l*H_, nullptr, 0,
             gbuf, H_, TOK_, H_, 2*H_, 2 /*sigmoid*/, 0.0f);
        gemm(comb, 2*H_, W_merge + (size_t)l*2*H_*H_, H_, nullptr, nullptr, 0,
             mbuf, H_, TOK_, H_, 2*H_, 0, 0.0f);
        combine_kernel<<<ew(TOK_*H_), 256, 0, stream>>>(gbuf, hf, hb, mbuf, x, TOK_*H_);
    }

    // ---- 3. decoder over all P patches ----
    build_full_kernel<<<ew(TOKP_*H_), 256, 0, stream>>>(mask_tok, pos_embed, full, TOKP_*H_);
    scatter_vis_kernel<<<ew(TOK_*H_), 256, 0, stream>>>(x, vis, full, TOK_*H_);
    gemm(full, H_, dec_W1, H_, dec_b1, nullptr, 0, hdec, H_, TOKP_, H_, H_, 3 /*gelu*/, 0.0f);
    gemm(hdec, H_, dec_W2, D_, dec_b2, nullptr, 0, (float*)d_out, D_, TOKP_, D_, H_, 0, 0.0f);
}
